// SAModule_76519137345577
// MI455X (gfx1250) — matrix operand order repaired and checked by reading, not compiled, not run on hardware
//
#include <hip/hip_runtime.h>
#include <hip/hip_bf16.h>

typedef __attribute__((ext_vector_type(16))) _Float16 v16h;
typedef __attribute__((ext_vector_type(8)))  float    v8f;

#define NPTS 16384
#define MCTR 4096
#define KNBR 64
#define R2C  0.04f
#define CAND_MAX 1024

// Pre-transposed f16 weight image: W1^T [64][96] | W2^T [64][64] | W3^T [128][64]
#define W1T_ELEMS (64 * 96)
#define W2T_ELEMS (64 * 64)
#define W3T_ELEMS (128 * 64)
#define WPRE_ELEMS (W1T_ELEMS + W2T_ELEMS + W3T_ELEMS)   // 18432 f16 = 36 KB

// ---------------------------------------------------------------------------
// Kernel 1: Farthest Point Sampling. Single 1024-thread workgroup; each thread
// owns 16 points (coords + running min-d2 in VGPRs). Per step: broadcast the
// current farthest point, update min-d2, wave32 shfl_xor argmax, then a 32-wave
// LDS reduction. 4096 dependent steps -> latency bound, keep everything in regs.
// ---------------------------------------------------------------------------
__global__ __launch_bounds__(1024)
void fps_kernel(const float* __restrict__ pos, int* __restrict__ idx)
{
    const int tid = threadIdx.x;
    __shared__ float sv[32];
    __shared__ int   si[32];
    __shared__ int   sfar;

    float px[16], py[16], pz[16], mind[16];
#pragma unroll
    for (int i = 0; i < 16; ++i) {
        int p = tid + i * 1024;
        px[i] = pos[p * 3 + 0];
        py[i] = pos[p * 3 + 1];
        pz[i] = pos[p * 3 + 2];
        mind[i] = 1e10f;
    }

    int far = 0;
    for (int m = 0; m < MCTR; ++m) {
        if (tid == 0) idx[m] = far;
        const float cx = pos[far * 3 + 0];   // uniform address -> broadcast load
        const float cy = pos[far * 3 + 1];
        const float cz = pos[far * 3 + 2];

        float bv = -1.0f; int bi = 0x7fffffff;
#pragma unroll
        for (int i = 0; i < 16; ++i) {
            float dx = px[i] - cx, dy = py[i] - cy, dz = pz[i] - cz;
            float d2 = dx * dx + dy * dy + dz * dz;
            mind[i] = fminf(mind[i], d2);
            int p = tid + i * 1024;
            if (mind[i] > bv || (mind[i] == bv && p < bi)) { bv = mind[i]; bi = p; }
        }
#pragma unroll
        for (int s = 16; s > 0; s >>= 1) {
            float ov = __shfl_xor(bv, s);
            int   oi = __shfl_xor(bi, s);
            if (ov > bv || (ov == bv && oi < bi)) { bv = ov; bi = oi; }
        }
        const int wid = tid >> 5;
        if ((tid & 31) == 0) { sv[wid] = bv; si[wid] = bi; }
        __syncthreads();
        if (tid < 32) {
            bv = sv[tid]; bi = si[tid];
#pragma unroll
            for (int s = 16; s > 0; s >>= 1) {
                float ov = __shfl_xor(bv, s);
                int   oi = __shfl_xor(bi, s);
                if (ov > bv || (ov == bv && oi < bi)) { bv = ov; bi = oi; }
            }
            if (tid == 0) sfar = bi;
        }
        __syncthreads();
        far = sfar;
    }
}

// ---------------------------------------------------------------------------
// Kernel 2: gather sampled positions / batch into the output tail.
// ---------------------------------------------------------------------------
__global__ void sample_kernel(const float* __restrict__ pos, const int* __restrict__ idx,
                              float* __restrict__ outPos, float* __restrict__ outBatch)
{
    int t = blockIdx.x * blockDim.x + threadIdx.x;
    if (t < MCTR) {
        int j = idx[t];
        outPos[t * 3 + 0] = pos[j * 3 + 0];
        outPos[t * 3 + 1] = pos[j * 3 + 1];
        outPos[t * 3 + 2] = pos[j * 3 + 2];
        outBatch[t] = 0.0f;   // single batch; int32 0 and f32 0.0 share a bit pattern
    }
}

// ---------------------------------------------------------------------------
// Kernel 3: one-time weight preparation: f32 -> f16, transpose to [n][k]
// (so WMMA B-fragment K-pairs are contiguous), zero-pad W1's K 67 -> 96.
// ---------------------------------------------------------------------------
__global__ void prep_weights(const float* __restrict__ W1, const float* __restrict__ W2,
                             const float* __restrict__ W3, _Float16* __restrict__ wPre)
{
    int t = blockIdx.x * blockDim.x + threadIdx.x;
    if (t < W1T_ELEMS) {
        int n = t / 96, k = t % 96;
        wPre[t] = (k < 67) ? (_Float16)W1[k * 64 + n] : (_Float16)0.0f;
    } else if (t < W1T_ELEMS + W2T_ELEMS) {
        int u = t - W1T_ELEMS;
        int n = u >> 6, k = u & 63;
        wPre[t] = (_Float16)W2[k * 64 + n];
    } else if (t < WPRE_ELEMS) {
        int u = t - (W1T_ELEMS + W2T_ELEMS);
        int n = u >> 6, k = u & 63;
        wPre[t] = (_Float16)W3[k * 128 + n];
    }
}

// ---------------------------------------------------------------------------
// Kernel 4: ball query. One 256-thread block per centroid. Candidates within
// R^2 are atomically appended to LDS; if more than K, take the K nearest via
// 64 rounds of LDS argmin (rare: expected ~35 in-radius points for this data).
// Neighbor *sets* are order-invariant under the final max-aggregation.
// ---------------------------------------------------------------------------
__global__ __launch_bounds__(256)
void radius_kernel(const float* __restrict__ pos, const int* __restrict__ idx,
                   int* __restrict__ nbr, int* __restrict__ cntOut)
{
    const int m = blockIdx.x;
    const int tid = threadIdx.x;
    __shared__ float cd2[CAND_MAX];
    __shared__ int   cid[CAND_MAX];
    __shared__ int   ccnt;
    __shared__ float rv[256];
    __shared__ int   ri[256];

    if (tid == 0) ccnt = 0;
    __syncthreads();

    const int ic = idx[m];
    const float cx = pos[ic * 3 + 0], cy = pos[ic * 3 + 1], cz = pos[ic * 3 + 2];

    for (int i = tid; i < NPTS; i += 256) {
        float dx = pos[i * 3 + 0] - cx;
        float dy = pos[i * 3 + 1] - cy;
        float dz = pos[i * 3 + 2] - cz;
        float d2 = dx * dx + dy * dy + dz * dz;
        if (d2 <= R2C) {
            int s = atomicAdd(&ccnt, 1);
            if (s < CAND_MAX) { cd2[s] = d2; cid[s] = i; }
        }
    }
    __syncthreads();

    int total = ccnt;
    if (total > CAND_MAX) total = CAND_MAX;

    if (total <= KNBR) {
        if (tid < total) nbr[m * KNBR + tid] = cid[tid];
        if (tid == 0) cntOut[m] = total;
    } else {
        for (int r = 0; r < KNBR; ++r) {
            float bv = 1e30f; int bi = -1;
            for (int t = tid; t < total; t += 256)
                if (cd2[t] < bv) { bv = cd2[t]; bi = t; }
            rv[tid] = bv; ri[tid] = bi;
            __syncthreads();
            for (int s = 128; s > 0; s >>= 1) {
                if (tid < s) {
                    if (rv[tid + s] < rv[tid]) { rv[tid] = rv[tid + s]; ri[tid] = ri[tid + s]; }
                }
                __syncthreads();
            }
            if (tid == 0) { nbr[m * KNBR + r] = cid[ri[0]]; cd2[ri[0]] = 1e30f; }
            __syncthreads();
        }
        if (tid == 0) cntOut[m] = KNBR;
    }
}

// ---------------------------------------------------------------------------
// WMMA fragment loaders (ISA 7.12.2 layouts, wave32).
//   A 16x32 f16: lane = 16h + m ; VGPR j holds K = base + 2(j&3) + 16*(j>=4) + 8h, +1
//   B 32x16 f16: lane = 16h + n ; VGPR j holds K = base + 2(j&3) + 16*(j>=4) + 8h, +1
//     (weights staged transposed in LDS: Wt[n][k] so pairs are contiguous)
//   D 16x16 f32: VGPR r -> row = 16*mt + r + 8h, col = 16*nt + (lane&15)
// ---------------------------------------------------------------------------
__device__ inline v16h load_a_frag(const _Float16* base, int row, int ld, int K0, int h)
{
    v16h a;
#pragma unroll
    for (int j = 0; j < 8; ++j) {
        int koff = ((j & 4) << 2) + 2 * (j & 3) + 8 * h;
        const _Float16* p = base + row * ld + K0 + koff;
        a[2 * j]     = p[0];
        a[2 * j + 1] = p[1];
    }
    return a;
}

__device__ inline v16h load_b_frag(const _Float16* baseT, int col, int ld, int K0, int h)
{
    v16h b;
#pragma unroll
    for (int j = 0; j < 8; ++j) {
        int k = K0 + ((j & 4) << 2) + 2 * (j & 3) + 8 * h;
        const _Float16* p = baseT + col * ld + k;
        b[2 * j]     = p[0];
        b[2 * j + 1] = p[1];
    }
    return b;
}

// ---------------------------------------------------------------------------
// Kernel 5: fused PointNetConv MLP + max aggregation. One workgroup (4 waves)
// per centroid. Weights arrive in LDS via GLOBAL_LOAD_ASYNC_TO_LDS_B128
// (ASYNCcnt-tracked, overlapped with the scattered H0 gather). All three
// GEMMs use v_wmma_f32_16x16x32_f16; layer-3 fuses bias+ReLU+mask+max.
// ---------------------------------------------------------------------------
__global__ __launch_bounds__(128)
void mlp_kernel(const float* __restrict__ x, const float* __restrict__ pos,
                const _Float16* __restrict__ wPre,
                const float* __restrict__ b1, const float* __restrict__ b2,
                const float* __restrict__ b3,
                const int* __restrict__ idx, const int* __restrict__ nbr,
                const int* __restrict__ cnt, float* __restrict__ out)
{
    const int m    = blockIdx.x;
    const int tid  = threadIdx.x;
    const int wid  = tid >> 5;
    const int lane = tid & 31;
    const int h    = lane >> 4;
    const int ln   = lane & 15;

    __shared__ _Float16 sW[WPRE_ELEMS];  // W1^T | W2^T | W3^T  (36 KB)
    __shared__ _Float16 sHA[64 * 96];    // H0 (ld 96), reused as H2 (ld 64)
    __shared__ _Float16 sHB[64 * 64];    // H1 (ld 64)

    const _Float16* sW1 = sW;
    const _Float16* sW2 = sW + W1T_ELEMS;
    const _Float16* sW3 = sW + W1T_ELEMS + W2T_ELEMS;

    // ---- async-copy the prepared f16 weight image into LDS -----------------
    // 36864 B / 16 B = 2304 B128 transfers; 128 lanes x 18 iterations.
    {
        unsigned lds_base = (unsigned)(uintptr_t)&sW[0];   // addr[31:0] = LDS offset
#pragma unroll
        for (int it = 0; it < (WPRE_ELEMS * 2 / 16) / 128; ++it) {
            unsigned off  = (unsigned)(tid + it * 128) * 16u;
            unsigned ldsa = lds_base + off;
            asm volatile("global_load_async_to_lds_b128 %0, %1, %2"
                         :
                         : "v"(ldsa), "v"(off), "s"(wPre)
                         : "memory");
        }
    }

    // ---- build H0 = [x_j | pos_j - pos_i | 0] while the copy is in flight --
    const int cvalid = cnt[m];
    const int ic = idx[m];
    const float cx = pos[ic * 3 + 0], cy = pos[ic * 3 + 1], cz = pos[ic * 3 + 2];

    for (int t = tid; t < 64 * 96; t += 128) {
        int slot = t / 96, cc = t % 96;
        float v = 0.0f;
        if (slot < cvalid && cc < 67) {
            int j = nbr[m * KNBR + slot];
            if (cc < 64) {
                v = x[j * 64 + cc];
            } else {
                int d = cc - 64;
                float cpi = (d == 0) ? cx : (d == 1) ? cy : cz;
                v = pos[j * 3 + d] - cpi;
            }
        }
        sHA[t] = (_Float16)v;
    }

    // every wave drains its own async loads, then the barrier makes all of
    // LDS (weights + H0) visible workgroup-wide
    asm volatile("s_wait_asynccnt 0" ::: "memory");
    __syncthreads();

    // ---- layer 1: H1 = relu(H0[64x96] @ W1p[96x64] + b1) -------------------
    {
        const int mt = wid;
        for (int nt = 0; nt < 4; ++nt) {
            v8f acc = {};
#pragma unroll
            for (int ks = 0; ks < 3; ++ks) {
                v16h a = load_a_frag(sHA, 16 * mt + ln, 96, 32 * ks, h);
                v16h b = load_b_frag(sW1, 16 * nt + ln, 96, 32 * ks, h);
                acc = __builtin_amdgcn_wmma_f32_16x16x32_f16(
                          false, a, false, b, (short)0, acc, false, false);
            }
            const int n = 16 * nt + ln;
            const float bias = b1[n];
#pragma unroll
            for (int r = 0; r < 8; ++r) {
                int row = 16 * mt + r + 8 * h;
                float v = acc[r] + bias;
                v = v > 0.0f ? v : 0.0f;
                sHB[row * 64 + n] = (_Float16)v;
            }
        }
    }
    __syncthreads();

    // ---- layer 2: H2 = relu(H1[64x64] @ W2[64x64] + b2) (into sHA, ld 64) --
    {
        const int mt = wid;
        for (int nt = 0; nt < 4; ++nt) {
            v8f acc = {};
#pragma unroll
            for (int ks = 0; ks < 2; ++ks) {
                v16h a = load_a_frag(sHB, 16 * mt + ln, 64, 32 * ks, h);
                v16h b = load_b_frag(sW2, 16 * nt + ln, 64, 32 * ks, h);
                acc = __builtin_amdgcn_wmma_f32_16x16x32_f16(
                          false, a, false, b, (short)0, acc, false, false);
            }
            const int n = 16 * nt + ln;
            const float bias = b2[n];
#pragma unroll
            for (int r = 0; r < 8; ++r) {
                int row = 16 * mt + r + 8 * h;
                float v = acc[r] + bias;
                v = v > 0.0f ? v : 0.0f;
                sHA[row * 64 + n] = (_Float16)v;
            }
        }
    }
    __syncthreads();

    // ---- layer 3: relu(H2 @ W3 + b3), masked max over valid neighbor rows --
    for (int t3 = 0; t3 < 2; ++t3) {
        const int nt = wid + 4 * t3;         // 8 column-tiles of 16 -> 128 cols
        const int n  = 16 * nt + ln;
        const float bias = b3[n];
        float colmax = -1e30f;
        for (int mt = 0; mt < 4; ++mt) {
            v8f acc = {};
#pragma unroll
            for (int ks = 0; ks < 2; ++ks) {
                v16h a = load_a_frag(sHA, 16 * mt + ln, 64, 32 * ks, h);
                v16h b = load_b_frag(sW3, n, 64, 32 * ks, h);
                acc = __builtin_amdgcn_wmma_f32_16x16x32_f16(
                          false, a, false, b, (short)0, acc, false, false);
            }
#pragma unroll
            for (int r = 0; r < 8; ++r) {
                int row = 16 * mt + r + 8 * h;   // neighbor slot
                if (row < cvalid) {
                    float v = acc[r] + bias;
                    v = v > 0.0f ? v : 0.0f;
                    colmax = fmaxf(colmax, v);
                }
            }
        }
        // combine the two lane-halves (rows split 8-apart between halves)
        colmax = fmaxf(colmax, __shfl_xor(colmax, 16));
        if (h == 0) out[m * 128 + n] = (cvalid > 0) ? colmax : 0.0f;
    }
}

// ---------------------------------------------------------------------------
extern "C" void kernel_launch(void* const* d_in, const int* in_sizes, int n_in,
                              void* d_out, int out_size, void* d_ws, size_t ws_size,
                              hipStream_t stream)
{
    (void)in_sizes; (void)n_in; (void)out_size; (void)ws_size;

    const float* x   = (const float*)d_in[0];
    const float* pos = (const float*)d_in[1];
    // d_in[2] = batch (all zeros, single cloud) -> unused
    const float* W1 = (const float*)d_in[3];
    const float* b1 = (const float*)d_in[4];
    const float* W2 = (const float*)d_in[5];
    const float* b2 = (const float*)d_in[6];
    const float* W3 = (const float*)d_in[7];
    const float* b3 = (const float*)d_in[8];

    float* out      = (float*)d_out;               // [M,128]
    float* outPos   = out + MCTR * 128;            // [M,3]
    float* outBatch = outPos + MCTR * 3;           // [M]

    int* idx = (int*)d_ws;                         // [M]
    int* cnt = idx + MCTR;                         // [M]
    int* nbr = cnt + MCTR;                         // [M,K]
    _Float16* wPre = (_Float16*)(nbr + MCTR * KNBR);  // 18432 f16, 16B-aligned

    fps_kernel   <<<1,    1024, 0, stream>>>(pos, idx);
    sample_kernel<<<16,   256,  0, stream>>>(pos, idx, outPos, outBatch);
    prep_weights <<<72,   256,  0, stream>>>(W1, W2, W3, wPre);
    radius_kernel<<<MCTR, 256,  0, stream>>>(pos, idx, nbr, cnt);
    mlp_kernel   <<<MCTR, 128,  0, stream>>>(x, pos, wPre, b1, b2, b3,
                                             idx, nbr, cnt, out);
}
